// RNN_50964081934372
// MI455X (gfx1250) — matrix operand order repaired and checked by reading, not compiled, not run on hardware
//
#include <hip/hip_runtime.h>
#include <math.h>

// Problem constants (from reference): fixed, so T is hardcoded (the scalar T
// input lives in device memory and cannot be read during graph capture).
#define BATCH   256
#define IN_SZ   256
#define HID     1024
#define OUT_SZ  256
#define T_STEPS 512

typedef __attribute__((ext_vector_type(16))) __bf16 bf16x16;
typedef __attribute__((ext_vector_type(8)))  float  floatx8;

union Frag {
    bf16x16 v;
    uint4   q[2];
};

// ---------------------------------------------------------------------------
// A-operand fragment: 16x32 (MxK) tile from row-major bf16 X[rows][K].
// CDNA5 16-bit A layout (wave32): lanes 0-15 hold M=lane, elements 0..7 =
// K kb+0..7, elements 8..15 = K kb+16..23; lanes 16-31 hold M=lane-16,
// elements 0..7 = K kb+8..15, elements 8..15 = K kb+24..31.
// => two contiguous 16-byte loads per lane.
// ---------------------------------------------------------------------------
__device__ __forceinline__ Frag load_a(const __bf16* __restrict__ X, int K,
                                       int mbase, int kbase, int lane) {
    Frag f;
    const int row  = mbase + (lane & 15);
    const int half = (lane >> 4) << 3;          // 0 or 8
    const __bf16* p = X + (size_t)row * K + kbase + half;
    f.q[0] = *reinterpret_cast<const uint4*>(p);        // K = kb+half .. +7
    f.q[1] = *reinterpret_cast<const uint4*>(p + 16);   // K = kb+half+16 .. +23
    return f;
}

// ---------------------------------------------------------------------------
// B-operand fragment from pre-packed weights. Packed layout per (nT,kT) tile:
// lane l holds 16 contiguous bf16 of row W[nbase + (l&15)]: k = kbase + (l>>4)*8 + 0..7, then + 16..23
// matching the CDNA5 32x16 (KxN) B layout exactly.
// ---------------------------------------------------------------------------
__device__ __forceinline__ Frag load_b(const __bf16* __restrict__ Wp,
                                       int Ktiles, int nT, int kT, int lane) {
    Frag f;
    const __bf16* p = Wp + (((size_t)nT * Ktiles + kT) * 32 + lane) * 16;
    f.q[0] = *reinterpret_cast<const uint4*>(p);
    f.q[1] = *reinterpret_cast<const uint4*>(p + 8);
    return f;
}

// ---------------------------------------------------------------------------
// Pack fp32 weights W[N][K] (row-major, K contiguous) into WMMA B layout bf16.
// ---------------------------------------------------------------------------
__global__ void __launch_bounds__(256)
pack_weight(const float* __restrict__ W, __bf16* __restrict__ Wp, int N, int K) {
    int idx = blockIdx.x * blockDim.x + threadIdx.x;
    if (idx >= N * K) return;
    int n = idx / K, k = idx - n * K;
    int nT = n >> 4, kT = k >> 5;
    int lane = (n & 15) + (((k >> 3) & 1) << 4);
    int e = (k & 7) | (((k >> 4) & 1) << 3);
    Wp[(((size_t)nT * (K >> 5) + kT) * 32 + lane) * 16 + e] = (__bf16)W[idx];
}

// ---------------------------------------------------------------------------
// Initial state: trajectory[0] = x_0; x_bf16 = bf16(x_0); h_bf16 = 0.
// ---------------------------------------------------------------------------
__global__ void __launch_bounds__(256)
init_state(const float* __restrict__ x0, __bf16* __restrict__ xb,
           __bf16* __restrict__ hb, float* __restrict__ out0) {
    int idx = blockIdx.x * blockDim.x + threadIdx.x;
    if (idx < BATCH * IN_SZ) {
        float v = x0[idx];
        out0[idx] = v;
        xb[idx]   = (__bf16)v;
    }
    if (idx < BATCH * HID) hb[idx] = (__bf16)0.0f;
}

// ---------------------------------------------------------------------------
// Hidden-state step: h_out = tanh(x·W_ih^T + h_in·W_hh^T + b_ih + b_hh).
// One wave per 16x32 output tile: 512 waves = 64 blocks x 8 waves.
// ---------------------------------------------------------------------------
__global__ void __launch_bounds__(256)
rnn_step_h(const __bf16* __restrict__ xb, const __bf16* __restrict__ hin,
           __bf16* __restrict__ hout,
           const float* __restrict__ b_ih, const float* __restrict__ b_hh,
           const __bf16* __restrict__ Wih, const __bf16* __restrict__ Whh) {
    const int lane  = threadIdx.x & 31;
    const int wid   = blockIdx.x * 8 + (threadIdx.x >> 5);
    const int mbase = (wid & 15) << 4;     // 16 batch rows
    const int nbase = (wid >> 4) << 5;     // 32 hidden cols
    const int nT0   = nbase >> 4, nT1 = nT0 + 1;

    floatx8 acc0 = {};
    floatx8 acc1 = {};

    // Input GEMM: K = IN_SZ (8 K-tiles of 32)
#pragma unroll 4
    for (int kT = 0; kT < IN_SZ / 32; ++kT) {
        Frag a  = load_a(xb, IN_SZ, mbase, kT << 5, lane);
        Frag b0 = load_b(Wih, IN_SZ / 32, nT0, kT, lane);
        Frag b1 = load_b(Wih, IN_SZ / 32, nT1, kT, lane);
        acc0 = __builtin_amdgcn_wmma_f32_16x16x32_bf16(false, a.v, false, b0.v,
                                                       (short)0, acc0, false, false);
        acc1 = __builtin_amdgcn_wmma_f32_16x16x32_bf16(false, a.v, false, b1.v,
                                                       (short)0, acc1, false, false);
    }

    // Recurrent GEMM: K = HID (32 K-tiles of 32)
#pragma unroll 4
    for (int kT = 0; kT < HID / 32; ++kT) {
        Frag a  = load_a(hin, HID, mbase, kT << 5, lane);
        Frag b0 = load_b(Whh, HID / 32, nT0, kT, lane);
        Frag b1 = load_b(Whh, HID / 32, nT1, kT, lane);
        acc0 = __builtin_amdgcn_wmma_f32_16x16x32_bf16(false, a.v, false, b0.v,
                                                       (short)0, acc0, false, false);
        acc1 = __builtin_amdgcn_wmma_f32_16x16x32_bf16(false, a.v, false, b1.v,
                                                       (short)0, acc1, false, false);
    }

    // Epilogue: C/D layout => col = nbase + t*16 + (lane&15); row = mbase +
    // vgpr + 8*(lane>>4). Bias + tanh in fp32, store bf16 for next step.
    const int c0 = nbase + (lane & 15);
    const int c1 = c0 + 16;
    const float bias0 = b_ih[c0] + b_hh[c0];
    const float bias1 = b_ih[c1] + b_hh[c1];
    const int rbase = mbase + ((lane >> 4) << 3);
#pragma unroll
    for (int v = 0; v < 8; ++v) {
        const size_t row = (size_t)(rbase + v);
        hout[row * HID + c0] = (__bf16)tanhf(acc0[v] + bias0);
        hout[row * HID + c1] = (__bf16)tanhf(acc1[v] + bias1);
    }
}

// ---------------------------------------------------------------------------
// Output step: y = h·W_fc^T + b_fc. Writes fp32 trajectory slice AND bf16 copy
// as next step's x. 128 waves = 16 blocks x 8 waves.
// ---------------------------------------------------------------------------
__global__ void __launch_bounds__(256)
rnn_step_y(const __bf16* __restrict__ hb, const __bf16* __restrict__ Wfc,
           const float* __restrict__ b_fc, float* __restrict__ out_t,
           __bf16* __restrict__ xb_next) {
    const int lane  = threadIdx.x & 31;
    const int wid   = blockIdx.x * 8 + (threadIdx.x >> 5);
    const int mbase = (wid & 15) << 4;
    const int nbase = (wid >> 4) << 5;     // 8 nGroups * 32 = 256 cols
    const int nT0   = nbase >> 4, nT1 = nT0 + 1;

    floatx8 acc0 = {};
    floatx8 acc1 = {};

#pragma unroll 4
    for (int kT = 0; kT < HID / 32; ++kT) {
        Frag a  = load_a(hb, HID, mbase, kT << 5, lane);
        Frag b0 = load_b(Wfc, HID / 32, nT0, kT, lane);
        Frag b1 = load_b(Wfc, HID / 32, nT1, kT, lane);
        acc0 = __builtin_amdgcn_wmma_f32_16x16x32_bf16(false, a.v, false, b0.v,
                                                       (short)0, acc0, false, false);
        acc1 = __builtin_amdgcn_wmma_f32_16x16x32_bf16(false, a.v, false, b1.v,
                                                       (short)0, acc1, false, false);
    }

    const int c0 = nbase + (lane & 15);
    const int c1 = c0 + 16;
    const float bias0 = b_fc[c0];
    const float bias1 = b_fc[c1];
    const int rbase = mbase + ((lane >> 4) << 3);
#pragma unroll
    for (int v = 0; v < 8; ++v) {
        const size_t row = (size_t)(rbase + v);
        const float y0 = acc0[v] + bias0;
        const float y1 = acc1[v] + bias1;
        out_t[row * OUT_SZ + c0]   = y0;
        out_t[row * OUT_SZ + c1]   = y1;
        xb_next[row * OUT_SZ + c0] = (__bf16)y0;
        xb_next[row * OUT_SZ + c1] = (__bf16)y1;
    }
}

// ---------------------------------------------------------------------------
// Host-side orchestration.
// ---------------------------------------------------------------------------
extern "C" void kernel_launch(void* const* d_in, const int* in_sizes, int n_in,
                              void* d_out, int out_size, void* d_ws, size_t ws_size,
                              hipStream_t stream) {
    const float* x_0  = (const float*)d_in[0];
    const float* W_ih = (const float*)d_in[1];
    const float* b_ih = (const float*)d_in[2];
    const float* W_hh = (const float*)d_in[3];
    const float* b_hh = (const float*)d_in[4];
    const float* W_fc = (const float*)d_in[5];
    const float* b_fc = (const float*)d_in[6];
    float* out = (float*)d_out;

    // Workspace carve-out (bf16 buffers, all sizes 256B-aligned).
    uint8_t* ws = (uint8_t*)d_ws;
    __bf16* Wih_p = (__bf16*)ws;  ws += (size_t)HID * IN_SZ * 2;      // 512 KB
    __bf16* Whh_p = (__bf16*)ws;  ws += (size_t)HID * HID * 2;        // 2 MB
    __bf16* Wfc_p = (__bf16*)ws;  ws += (size_t)OUT_SZ * HID * 2;     // 512 KB
    __bf16* xb    = (__bf16*)ws;  ws += (size_t)BATCH * IN_SZ * 2;    // 128 KB
    __bf16* hb0   = (__bf16*)ws;  ws += (size_t)BATCH * HID * 2;      // 512 KB
    __bf16* hb1   = (__bf16*)ws;  ws += (size_t)BATCH * HID * 2;      // 512 KB

    // One-time (per call) weight packing into WMMA B layout + state init.
    pack_weight<<<(HID * IN_SZ + 255) / 256, 256, 0, stream>>>(W_ih, Wih_p, HID, IN_SZ);
    pack_weight<<<(HID * HID + 255) / 256, 256, 0, stream>>>(W_hh, Whh_p, HID, HID);
    pack_weight<<<(OUT_SZ * HID + 255) / 256, 256, 0, stream>>>(W_fc, Wfc_p, OUT_SZ, HID);
    init_state<<<(BATCH * HID + 255) / 256, 256, 0, stream>>>(x_0, xb, hb0, out);

    // Autoregressive rollout: trajectory[t] for t = 1..T-1.
    for (int t = 1; t < T_STEPS; ++t) {
        const __bf16* hin  = (t & 1) ? hb0 : hb1;   // t=1 reads zeroed hb0
        __bf16*       hout = (t & 1) ? hb1 : hb0;
        rnn_step_h<<<64, 256, 0, stream>>>(xb, hin, hout, b_ih, b_hh, Wih_p, Whh_p);
        rnn_step_y<<<16, 256, 0, stream>>>(hout, Wfc_p, b_fc,
                                           out + (size_t)t * BATCH * OUT_SZ, xb);
    }
}
